// Seq2VecRNN2D_29343216566780
// MI455X (gfx1250) — compile-verified, matrix operand order checked by reading, not run on hardware
//
#include <hip/hip_runtime.h>

// ---------------- problem constants ----------------
#define DINLINE __device__ __forceinline__
static constexpr int Bb   = 8;      // batch
static constexpr int T    = 8;      // time steps
static constexpr int HIDc = 64;     // hidden channels
static constexpr int HW   = 1024;   // 32*32
static constexpr int G    = 256;    // 4*HID gate channels (GEMM M)
static constexpr int CIN  = 128;    // layer input channels (both layers)
static constexpr int KIH  = CIN * 9;        // 1152
static constexpr int KHH  = HIDc * 9;       // 576
static constexpr int KTOT = KIH + KHH;      // 1728 (54 * 32)
static constexpr int NK   = KTOT / 32;      // 54 K-steps
static constexpr int NTOT = Bb * HW;        // 8192 (GEMM N)
static constexpr int OUTC = 64;
static constexpr int LDK  = 34;             // padded LDS K stride (u16 units)

typedef __attribute__((ext_vector_type(8)))  float        v8f;
typedef __attribute__((ext_vector_type(8)))  unsigned int v8u;
typedef __attribute__((ext_vector_type(16))) __bf16       v16bf;

DINLINE unsigned short f2bf(float f) {
  unsigned int u = __builtin_bit_cast(unsigned int, f);
  u += 0x7FFFu + ((u >> 16) & 1u);              // round-to-nearest-even
  return (unsigned short)(u >> 16);
}
DINLINE float bf2f(unsigned short h) {
  return __builtin_bit_cast(float, ((unsigned int)h) << 16);
}
DINLINE float sigm(float x) { return 1.0f / (1.0f + __expf(-x)); }

// ---------------- utility kernels ----------------
__global__ void k_zero_u32(unsigned int* __restrict__ p, int n) {
  int i = blockIdx.x * blockDim.x + threadIdx.x;
  if (i < n) p[i] = 0u;
}

// features [B][T][CIN][HW] (f32) -> featb [T][B][CIN][HW] (bf16)
__global__ void k_feat_to_bf16(const float* __restrict__ src,
                               unsigned short* __restrict__ dst) {
  int idx = blockIdx.x * blockDim.x + threadIdx.x;   // < T*B*CIN*HW = 2^23
  int p = idx & (HW - 1);
  int c = (idx >> 10) & (CIN - 1);
  int b = (idx >> 17) & (Bb - 1);
  int t = idx >> 20;
  dst[idx] = f2bf(src[((b * T + t) * CIN + c) * HW + p]);
}

// pack W_ih [G][CIN][3][3] + W_hh [G][HID][3][3] -> Apack [G][KTOT] bf16
__global__ void k_pack_w(const float* __restrict__ wih,
                         const float* __restrict__ whh,
                         unsigned short* __restrict__ ap) {
  int idx = blockIdx.x * blockDim.x + threadIdx.x;   // < G*KTOT
  int m = idx / KTOT;
  int k = idx - m * KTOT;
  float v;
  if (k < KIH) {
    int c = k / 9, r = k - c * 9;
    v = wih[(m * CIN + c) * 9 + r];
  } else {
    int kk = k - KIH;
    int c = kk / 9, r = kk - c * 9;
    v = whh[(m * HIDc + c) * 9 + r];
  }
  ap[idx] = f2bf(v);
}

// ---------------- WMMA implicit-GEMM gate kernel ----------------
// gates[M=256][N=8192] = Apack[M][K] x B[K][N], K enumerates (c,dy,dx) taps of
// x (ih part) and h_prev (hh part) with SAME zero padding.
// Block: 128 threads = 4 wave32; block tile 64(M) x 128(N); wave tile 64 x 32.
// Double-buffered K pipeline; weight tile staged via async global->LDS b128.
__global__ __launch_bounds__(128) void k_gemm_gates(
    const unsigned short* __restrict__ Apack,   // [G][KTOT] bf16
    const unsigned short* __restrict__ xin,     // [B][CIN][HW] bf16 (time t)
    const unsigned short* __restrict__ hprev,   // [B][HID][HW] bf16
    float* __restrict__ gates)                  // [G][NTOT] f32
{
  __shared__ unsigned short As[2][64 * LDK];    // [stage][m][k]
  __shared__ unsigned short Bs[2][128 * LDK];   // [stage][n][k]
  __shared__ int           kOff[KTOT];          // c*HW for tap k
  __shared__ unsigned char kSel[KTOT];          // bits0-1: dx+1, 2-3: dy+1, 4: hh

  const int tid  = threadIdx.x;
  const int lane = tid & 31;
  const int wave = tid >> 5;
  const int wn   = wave * 32;                   // wave N offset in block tile
  const int mBlock = blockIdx.x * 64;
  const int nBlock = blockIdx.y * 128;

  // ---- K decode table (removes /9,%9 from the hot gather) ----
  for (int kg = tid; kg < KTOT; kg += 128) {
    bool hh = kg >= KIH;
    int kl = hh ? kg - KIH : kg;
    int c = kl / 9, r = kl - c * 9;
    kOff[kg] = c * HW;
    kSel[kg] = (unsigned char)(((r / 3) << 2) | (r % 3) | (hh ? 16 : 0));
  }

  // ---- async weight-tile stage: 64 rows x 4 x b128 chunks ----
  auto fillA = [&](int s, int k0) {
    for (int task = tid; task < 64 * 4; task += 128) {
      int m = task >> 2, kc = task & 3;
      unsigned lds = (unsigned)(size_t)(&As[s][m * LDK + kc * 8]);
      const unsigned short* g =
          Apack + (size_t)(mBlock + m) * KTOT + k0 + kc * 8;
      asm volatile("global_load_async_to_lds_b128 %0, %1, off"
                   :: "v"(lds), "v"(g) : "memory");
    }
  };

  // ---- im2col gather: each thread owns one K column, 32 N rows ----
  auto fillB = [&](int s, int k0) {
    const int kk = tid & 31;
    const int kg = k0 + kk;
    const int off = kOff[kg];
    const unsigned dd = kSel[kg];
    const int dy = (int)((dd >> 2) & 3) - 1;
    const int dx = (int)(dd & 3) - 1;
    const bool ishh = (dd & 16) != 0;
    const unsigned short* src = ishh ? hprev : xin;
    const int bScale = ishh ? HIDc * HW : CIN * HW;
    const int n0 = tid >> 5;
#pragma unroll 4
    for (int j = 0; j < 32; ++j) {
      int n = n0 + j * 4;
      int ng = nBlock + n;
      int b = ng >> 10, p = ng & 1023;
      int y = p >> 5, x = p & 31;
      int yy = y + dy, xx = x + dx;
      unsigned short val = 0;
      if ((unsigned)yy < 32u && (unsigned)xx < 32u)
        val = src[b * bScale + off + (yy << 5) + xx];
      Bs[s][n * LDK + kk] = val;
    }
  };

  v8f acc[4][2];
  {
    v8f zero = {};
#pragma unroll
    for (int a = 0; a < 4; ++a)
#pragma unroll
      for (int b2 = 0; b2 < 2; ++b2) acc[a][b2] = zero;
  }

  // ---- fragment build + WMMA on one stage ----
  auto compute = [&](int s) {
    const int half = (lane >> 4) << 3;          // lanes16-31 take K+8
    const int l16  = lane & 15;
    v8u af4[4], bfr[2];
#pragma unroll
    for (int mt = 0; mt < 4; ++mt) {
      const unsigned int* row =
          (const unsigned int*)(&As[s][(mt * 16 + l16) * LDK]);
#pragma unroll
      for (int r = 0; r < 8; ++r) {
        int kb = ((r & 4) << 2) + half + ((r & 3) << 1);  // K pair base
        af4[mt][r] = row[kb >> 1];
      }
    }
#pragma unroll
    for (int nt = 0; nt < 2; ++nt) {
      const unsigned int* row =
          (const unsigned int*)(&Bs[s][(wn + nt * 16 + l16) * LDK]);
#pragma unroll
      for (int r = 0; r < 8; ++r) {
        int kb = ((r & 4) << 2) + half + ((r & 3) << 1);
        bfr[nt][r] = row[kb >> 1];
      }
    }
#pragma unroll
    for (int mt = 0; mt < 4; ++mt)
#pragma unroll
      for (int nt = 0; nt < 2; ++nt)
        acc[mt][nt] = __builtin_amdgcn_wmma_f32_16x16x32_bf16(
            false, __builtin_bit_cast(v16bf, af4[mt]),
            false, __builtin_bit_cast(v16bf, bfr[nt]),
            (short)0, acc[mt][nt], false, false);
  };

  // ---- double-buffered pipeline: one barrier per K-step ----
  fillA(0, 0);
  __syncthreads();                 // decode table visible
  fillB(0, 0);
  asm volatile("s_wait_asynccnt 0" ::: "memory");
  __syncthreads();
  for (int it = 0; it < NK; ++it) {
    int cur = it & 1, nxt = cur ^ 1;
    if (it + 1 < NK) {
      fillA(nxt, (it + 1) * 32);   // async, overlaps with WMMA below
      fillB(nxt, (it + 1) * 32);
    }
    compute(cur);
    asm volatile("s_wait_asynccnt 0" ::: "memory");
    __syncthreads();
  }

  // ---- store D per 32-bit C/D layout: VGPR v -> M = v + 8*(lane/16) ----
  const int half = lane >> 4;
  const int l16  = lane & 15;
#pragma unroll
  for (int mt = 0; mt < 4; ++mt)
#pragma unroll
    for (int nt = 0; nt < 2; ++nt) {
      int n = nBlock + wn + nt * 16 + l16;
#pragma unroll
      for (int v = 0; v < 8; ++v) {
        int m = mBlock + mt * 16 + half * 8 + v;
        gates[m * NTOT + n] = acc[mt][nt][v];
      }
    }
}

// ---------------- pointwise LSTM cell ----------------
__global__ void k_cell(const float* __restrict__ gates,         // [G][NTOT]
                       const float* __restrict__ bih,
                       const float* __restrict__ bhh,
                       float* __restrict__ cbuf,                // [B][HID][HW]
                       unsigned short* __restrict__ hbuf,       // [B][HID][HW] bf16
                       unsigned short* __restrict__ hout)       // concat dest
{
  int idx = blockIdx.x * blockDim.x + threadIdx.x;   // < B*HID*HW = 2^19
  int p   = idx & 1023;
  int hid = (idx >> 10) & 63;
  int b   = idx >> 16;
  int n   = (b << 10) + p;
  float gi = gates[hid * NTOT + n]              + bih[hid]            + bhh[hid];
  float gf = gates[(HIDc + hid) * NTOT + n]     + bih[HIDc + hid]     + bhh[HIDc + hid];
  float gg = gates[(2 * HIDc + hid) * NTOT + n] + bih[2 * HIDc + hid] + bhh[2 * HIDc + hid];
  float go = gates[(3 * HIDc + hid) * NTOT + n] + bih[3 * HIDc + hid] + bhh[3 * HIDc + hid];
  float c  = cbuf[idx];
  float cn = sigm(gf) * c + sigm(gi) * tanhf(gg);
  float hn = sigm(go) * tanhf(cn);
  cbuf[idx] = cn;
  unsigned short hb = f2bf(hn);
  hbuf[idx] = hb;
  hout[(b * (2 * HIDc) + hid) * HW + p] = hb;
}

// ---------------- final 1x1 conv + ReLU ----------------
__global__ void k_out(const unsigned short* __restrict__ last,  // [B][128][HW] bf16
                      const float* __restrict__ wout,           // [64][128]
                      const float* __restrict__ bout,           // [64]
                      float* __restrict__ out)                  // [B][64][HW]
{
  int idx = blockIdx.x * blockDim.x + threadIdx.x;   // < B*OUTC*HW = 2^19
  int p = idx & 1023;
  int o = (idx >> 10) & 63;
  int b = idx >> 16;
  float acc = bout[o];
  const unsigned short* src = last + (size_t)(b * (2 * HIDc)) * HW + p;
  const float* wr = wout + o * (2 * HIDc);
#pragma unroll 8
  for (int c = 0; c < 2 * HIDc; ++c) acc += bf2f(src[c * HW]) * wr[c];
  out[idx] = fmaxf(acc, 0.0f);
}

// ---------------- host orchestration ----------------
extern "C" void kernel_launch(void* const* d_in, const int* in_sizes, int n_in,
                              void* d_out, int out_size, void* d_ws, size_t ws_size,
                              hipStream_t stream) {
  (void)in_sizes; (void)n_in; (void)out_size; (void)ws_size;
  const float* features = (const float*)d_in[0];
  // set order: (layer0,f), (layer0,r), (layer1,f), (layer1,r)
  const float* wih[4] = {(const float*)d_in[1],  (const float*)d_in[5],
                         (const float*)d_in[9],  (const float*)d_in[13]};
  const float* whh[4] = {(const float*)d_in[2],  (const float*)d_in[6],
                         (const float*)d_in[10], (const float*)d_in[14]};
  const float* bih[4] = {(const float*)d_in[3],  (const float*)d_in[7],
                         (const float*)d_in[11], (const float*)d_in[15]};
  const float* bhh[4] = {(const float*)d_in[4],  (const float*)d_in[8],
                         (const float*)d_in[12], (const float*)d_in[16]};
  const float* wout = (const float*)d_in[17];
  const float* bout = (const float*)d_in[18];
  float* out = (float*)d_out;

  char* ws = (char*)d_ws;
  size_t off = 0;
  auto take = [&](size_t bytes) -> char* {
    char* p = ws + off;
    off += (bytes + 255) & ~(size_t)255;
    return p;
  };
  const size_t ACT = (size_t)T * Bb * CIN * HW;                 // 2^23 elems
  unsigned short* featb = (unsigned short*)take(ACT * 2);       // layer0 input
  unsigned short* xs1   = (unsigned short*)take(ACT * 2);       // layer1 input
  unsigned short* hs2   = (unsigned short*)take(ACT * 2);       // layer1 output
  unsigned short* apk[4];
  for (int i = 0; i < 4; ++i) apk[i] = (unsigned short*)take((size_t)G * KTOT * 2);
  unsigned short* hbuf = (unsigned short*)take((size_t)Bb * HIDc * HW * 2);
  float* cbuf  = (float*)take((size_t)Bb * HIDc * HW * 4);
  float* gates = (float*)take((size_t)G * NTOT * 4);

  k_feat_to_bf16<<<(int)(ACT / 256), 256, 0, stream>>>(features, featb);
  for (int i = 0; i < 4; ++i)
    k_pack_w<<<(G * KTOT) / 256, 256, 0, stream>>>(wih[i], whh[i], apk[i]);

  const dim3 ggrid(4, 64, 1);            // M tiles x N tiles
  const int NELEM = Bb * HIDc * HW;      // 524288

  for (int set = 0; set < 4; ++set) {
    int layer = set >> 1;
    int dir   = set & 1;                 // 0 = forward, 1 = reverse
    const unsigned short* xin_base  = (layer == 0) ? featb : xs1;
    unsigned short*       hout_base = (layer == 0) ? xs1   : hs2;
    k_zero_u32<<<(NELEM / 2) / 256, 256, 0, stream>>>((unsigned int*)hbuf, NELEM / 2);
    k_zero_u32<<<NELEM / 256, 256, 0, stream>>>((unsigned int*)cbuf, NELEM);
    for (int s = 0; s < T; ++s) {
      int t = dir ? (T - 1 - s) : s;
      const unsigned short* xin = xin_base + (size_t)t * Bb * CIN * HW;
      k_gemm_gates<<<ggrid, 128, 0, stream>>>(apk[set], xin, hbuf, gates);
      unsigned short* hout =
          hout_base + (size_t)t * Bb * (2 * HIDc) * HW + (size_t)dir * HIDc * HW;
      k_cell<<<NELEM / 256, 256, 0, stream>>>(gates, bih[set], bhh[set], cbuf, hbuf, hout);
    }
  }
  const unsigned short* last = hs2 + (size_t)(T - 1) * Bb * (2 * HIDc) * HW;
  k_out<<<NELEM / 256, 256, 0, stream>>>(last, wout, bout, out);
}
